// SparseLyingConv_9826885173720
// MI455X (gfx1250) — matrix operand order repaired and checked by reading, not compiled, not run on hardware
//
#include <hip/hip_runtime.h>
#include <math.h>

#define NV   50000
#define NE   800000
#define IN_F 128
#define OUT_F 128
#define KE   256   // 2*IN_F for the edge MLP
#define ET   64    // edges per block in sheaf kernel (4 x 16-row tiles)

typedef __attribute__((ext_vector_type(16))) __bf16 v16bf;
typedef __attribute__((ext_vector_type(8)))  float  v8f;

union AFrag { v16bf v; uint4 q[2]; };

// ---------------- helper kernels ----------------

__global__ void cvt_bf16_kernel(const float* __restrict__ src,
                                __bf16* __restrict__ dst, int n) {
    int i = blockIdx.x * blockDim.x + threadIdx.x;
    if (i < n) dst[i] = (__bf16)src[i];
}

// wT[n*K + k] = (bf16) w[k*N + n]   (row-major [K,N] -> col-major [N,K])
__global__ void transpose_w_kernel(const float* __restrict__ w,
                                   __bf16* __restrict__ wT, int K, int N) {
    int i = blockIdx.x * blockDim.x + threadIdx.x;
    if (i >= K * N) return;
    int n = i % N, k = i / N;
    wT[(size_t)n * K + k] = (__bf16)w[(size_t)k * N + n];
}

__global__ void deg_fill_kernel(float* __restrict__ deg) {
    int i = blockIdx.x * blockDim.x + threadIdx.x;
    if (i < NV) deg[i] = 1.0f;   // self-loop
}

__global__ void deg_add_kernel(const int* __restrict__ src, float* __restrict__ deg) {
    int e = blockIdx.x * blockDim.x + threadIdx.x;
    if (e < NE) unsafeAtomicAdd(&deg[src[e]], 1.0f);
}

__global__ void dinv_kernel(const float* __restrict__ deg, float* __restrict__ dinv) {
    int i = blockIdx.x * blockDim.x + threadIdx.x;
    if (i < NV) dinv[i] = __frsqrt_rn(deg[i]);
}

// out[v,c] = dinv[v]^2 * x_lin[v,c] + bias[c]
__global__ void out_init_kernel(const float* __restrict__ xlin,
                                const float* __restrict__ dinv,
                                const float* __restrict__ bias,
                                float* __restrict__ out) {
    int i = blockIdx.x * blockDim.x + threadIdx.x;
    if (i >= NV * OUT_F) return;
    int v = i >> 7, c = i & 127;
    float di = dinv[v];
    out[i] = di * di * xlin[i] + bias[c];
}

// ---------------- x_lin = x @ W_lin via WMMA bf16 ----------------
// grid.x = NV/16 row tiles, 256 threads = 8 waves, wave w -> N-tile w
__global__ void xlin_gemm_kernel(const __bf16* __restrict__ xb,    // [NV, IN_F]
                                 const __bf16* __restrict__ wlinT, // [OUT_F, IN_F]
                                 float* __restrict__ xlin) {       // [NV, OUT_F]
    __shared__ __attribute__((aligned(16))) __bf16 lds[16 * IN_F];
    const int tile = blockIdx.x;
    const int tid  = threadIdx.x;

    // stage 16 rows x 128 bf16 = 256 x 16B chunks, one per thread
    {
        int row = tid >> 4;
        int ofs = (tid & 15) * 8;
        *(uint4*)(&lds[row * IN_F + ofs]) =
            *(const uint4*)(xb + (size_t)(tile * 16 + row) * IN_F + ofs);
    }
    __syncthreads();

    const int wave = tid >> 5, lane = tid & 31;
    const int hi   = lane >> 4, lrow = lane & 15;
    const int col  = (wave << 4) + lrow;   // B fragment column / output column

    v8f acc = {};
    #pragma unroll
    for (int k0 = 0; k0 < IN_F; k0 += 32) {
        AFrag a, b;
        a.q[0] = *(const uint4*)(&lds[lrow * IN_F + k0 + 8 * hi]);
        a.q[1] = *(const uint4*)(&lds[lrow * IN_F + k0 + 16 + 8 * hi]);
        const __bf16* bp = wlinT + (size_t)col * IN_F + k0 + 8 * hi;
        b.q[0] = *(const uint4*)(bp);
        b.q[1] = *(const uint4*)(bp + 16);
        acc = __builtin_amdgcn_wmma_f32_16x16x32_bf16(
                  false, a.v, false, b.v, (short)0, acc, false, false);
    }
    const int mbase = tile * 16 + 8 * hi;
    #pragma unroll
    for (int r = 0; r < 8; ++r)
        xlin[(size_t)(mbase + r) * OUT_F + col] = acc[r];
}

// ------- sheaf_w = tanh([x[src],x[dst]] @ W_edge + b) + fused scatter -------
// grid.x = NE/ET edge blocks of 64 edges, 256 threads = 8 waves,
// wave w -> N-tile w. B fragments (K=256 slice of W_edge^T) are loaded ONCE
// into registers and reused across the 4 edge tiles; tiles are processed in
// pairs with two independent accumulators for WMMA ILP.
__global__ void sheaf_gemm_kernel(const __bf16* __restrict__ xb,      // [NV, IN_F]
                                  const __bf16* __restrict__ wedgeT,  // [OUT_F, KE]
                                  const float* __restrict__ b_edge,   // [OUT_F]
                                  const int* __restrict__ src,
                                  const int* __restrict__ dst,
                                  const float* __restrict__ dinv,
                                  const float* __restrict__ xlin,     // [NV, OUT_F]
                                  float* __restrict__ out,            // [NV, OUT_F]
                                  float* __restrict__ sheaf) {        // [NE, OUT_F]
    __shared__ __attribute__((aligned(16))) __bf16 lds[ET * KE];  // 32 KB
    __shared__ int   s_src[ET], s_dst[ET];
    __shared__ float s_norm[ET];

    const int blk = blockIdx.x;          // 64-edge block
    const int tid = threadIdx.x;

    if (tid < ET) {
        int e = blk * ET + tid;
        int s = src[e], d = dst[e];
        s_src[tid] = s;  s_dst[tid] = d;
        s_norm[tid] = dinv[s] * dinv[d];
    }
    __syncthreads();

    // stage 64 rows x 256 bf16 = 2048 x 16B chunks -> 8 per thread (gather)
    #pragma unroll
    for (int c = tid; c < ET * KE / 8; c += 256) {
        int row = c >> 5;              // 32 chunks per row
        int ofs = (c & 31) * 8;        // element offset in concat(x[src],x[dst])
        const __bf16* base = (ofs < IN_F)
            ? (xb + (size_t)s_src[row] * IN_F + ofs)
            : (xb + (size_t)s_dst[row] * IN_F + (ofs - IN_F));
        *(uint4*)(&lds[row * KE + ofs]) = *(const uint4*)base;
    }
    __syncthreads();

    const int wave = tid >> 5, lane = tid & 31;
    const int hi   = lane >> 4, lrow = lane & 15;
    const int col  = (wave << 4) + lrow;

    // load all 8 B fragments once (64 VGPRs), reused for all 4 edge tiles
    AFrag b[8];
    #pragma unroll
    for (int ks = 0; ks < 8; ++ks) {
        const __bf16* bp = wedgeT + (size_t)col * KE + ks * 32 + 8 * hi;
        b[ks].q[0] = *(const uint4*)(bp);
        b[ks].q[1] = *(const uint4*)(bp + 16);
    }
    const float bn = b_edge[col];

    #pragma unroll
    for (int p = 0; p < 2; ++p) {               // tile pairs (0,1) and (2,3)
        const int t0 = 2 * p, t1 = 2 * p + 1;
        v8f acc0 = {}, acc1 = {};
        #pragma unroll
        for (int ks = 0; ks < 8; ++ks) {
            const int k0 = ks * 32;
            AFrag a0, a1;
            a0.q[0] = *(const uint4*)(&lds[(t0 * 16 + lrow) * KE + k0 + 8 * hi]);
            a0.q[1] = *(const uint4*)(&lds[(t0 * 16 + lrow) * KE + k0 + 16 + 8 * hi]);
            a1.q[0] = *(const uint4*)(&lds[(t1 * 16 + lrow) * KE + k0 + 8 * hi]);
            a1.q[1] = *(const uint4*)(&lds[(t1 * 16 + lrow) * KE + k0 + 16 + 8 * hi]);
            acc0 = __builtin_amdgcn_wmma_f32_16x16x32_bf16(
                       false, a0.v, false, b[ks].v, (short)0, acc0, false, false);
            acc1 = __builtin_amdgcn_wmma_f32_16x16x32_bf16(
                       false, a1.v, false, b[ks].v, (short)0, acc1, false, false);
        }
        // epilogue for both tiles: bias + tanh, write sheaf, scatter messages
        #pragma unroll
        for (int r = 0; r < 8; ++r) {
            int m0 = t0 * 16 + r + 8 * hi;       // edge row inside 64-edge block
            int m1 = t1 * 16 + r + 8 * hi;
            int e0 = blk * ET + m0;
            int e1 = blk * ET + m1;
            float s0 = tanhf(acc0[r] + bn);
            float s1 = tanhf(acc1[r] + bn);
            sheaf[(size_t)e0 * OUT_F + col] = s0;
            sheaf[(size_t)e1 * OUT_F + col] = s1;
            float msg0 = s_norm[m0] * s0 * xlin[(size_t)s_dst[m0] * OUT_F + col];
            float msg1 = s_norm[m1] * s1 * xlin[(size_t)s_dst[m1] * OUT_F + col];
            unsafeAtomicAdd(&out[(size_t)s_src[m0] * OUT_F + col], msg0);
            unsafeAtomicAdd(&out[(size_t)s_src[m1] * OUT_F + col], msg1);
        }
    }
}

// ---------------- launch ----------------

static inline size_t al256(size_t x) { return (x + 255) & ~(size_t)255; }

extern "C" void kernel_launch(void* const* d_in, const int* in_sizes, int n_in,
                              void* d_out, int out_size, void* d_ws, size_t ws_size,
                              hipStream_t stream) {
    const float* x      = (const float*)d_in[0];
    const int*   ei     = (const int*)d_in[1];     // [2, NE]
    const float* W_edge = (const float*)d_in[2];   // [KE, OUT_F]
    const float* b_edge = (const float*)d_in[3];
    const float* W_lin  = (const float*)d_in[4];   // [IN_F, OUT_F]
    const float* bias   = (const float*)d_in[5];

    const int* src = ei;
    const int* dst = ei + NE;

    float* out   = (float*)d_out;                       // [NV, OUT_F]
    float* sheaf = (float*)d_out + (size_t)NV * OUT_F;  // [NE, OUT_F]

    // workspace layout
    char* ws = (char*)d_ws;
    size_t o = 0;
    __bf16* xb     = (__bf16*)(ws + o); o = al256(o + (size_t)NV * IN_F * sizeof(__bf16));
    __bf16* wlinT  = (__bf16*)(ws + o); o = al256(o + (size_t)OUT_F * IN_F * sizeof(__bf16));
    __bf16* wedgeT = (__bf16*)(ws + o); o = al256(o + (size_t)OUT_F * KE * sizeof(__bf16));
    float*  xlin   = (float*)(ws + o);  o = al256(o + (size_t)NV * OUT_F * sizeof(float));
    float*  deg    = (float*)(ws + o);  o = al256(o + (size_t)NV * sizeof(float));
    float*  dinv   = (float*)(ws + o);  o = al256(o + (size_t)NV * sizeof(float));
    (void)ws_size; (void)in_sizes; (void)n_in; (void)out_size;

    const int T = 256;

    // precision conversion + weight transposes
    cvt_bf16_kernel<<<(NV * IN_F + T - 1) / T, T, 0, stream>>>(x, xb, NV * IN_F);
    transpose_w_kernel<<<(IN_F * OUT_F + T - 1) / T, T, 0, stream>>>(W_lin, wlinT, IN_F, OUT_F);
    transpose_w_kernel<<<(KE * OUT_F + T - 1) / T, T, 0, stream>>>(W_edge, wedgeT, KE, OUT_F);

    // degree -> D^{-1/2}
    deg_fill_kernel<<<(NV + T - 1) / T, T, 0, stream>>>(deg);
    deg_add_kernel<<<(NE + T - 1) / T, T, 0, stream>>>(src, deg);
    dinv_kernel<<<(NV + T - 1) / T, T, 0, stream>>>(deg, dinv);

    // x_lin = x @ W_lin   (WMMA)
    xlin_gemm_kernel<<<NV / 16, T, 0, stream>>>(xb, wlinT, xlin);

    // out = diag * x_lin + bias
    out_init_kernel<<<(NV * OUT_F + T - 1) / T, T, 0, stream>>>(xlin, dinv, bias, out);

    // edge MLP (WMMA, B cached in registers, dual accumulators) + fused scatter
    sheaf_gemm_kernel<<<NE / ET, T, 0, stream>>>(xb, wedgeT, b_edge, src, dst,
                                                 dinv, xlin, out, sheaf);
}
